// Transformer_2697239461947
// MI455X (gfx1250) — matrix operand order repaired and checked by reading, not compiled, not run on hardware
//
#include <hip/hip_runtime.h>
#include <math.h>
#include <stdint.h>

#define USE_ASYNC_LDS 1

// ---------------------------------------------------------------------------
// Types for CDNA5 WMMA (wave32): v_wmma_f32_16x16x32_bf16
// ---------------------------------------------------------------------------
typedef __bf16 bf16_t;
typedef __attribute__((ext_vector_type(16))) __bf16 v16bf;
typedef __attribute__((ext_vector_type(8)))  __bf16 v8bf;
typedef __attribute__((ext_vector_type(2)))  __bf16 v2bf;
typedef __attribute__((ext_vector_type(8)))  float  v8f;

static __device__ __forceinline__ v8f wmma_bf16(v16bf a, v16bf b, v8f c) {
  // (neg_a, A, neg_b, B, c_mod, C, reuse_a, reuse_b)
  return __builtin_amdgcn_wmma_f32_16x16x32_bf16(false, a, false, b,
                                                 (short)0, c, false, false);
}

// Within each 32-element K block, swap chunks [8..15] <-> [16..23] so that a
// lane's 16 A-matrix values (K {0..7,16..23} for lanes 0-15, K {8..15,24..31}
// for lanes 16-31) become one contiguous 32B region in LDS.
static __device__ __forceinline__ int permk(int c) {
  return (c & ~24) | ((c & 8) << 1) | ((c & 16) >> 1);
}

// XOR-swizzle 32B units by (row & 3) inside a 64-element (128B) LDS row.
// Preserves e mod 16, so 16B/32B alignment and permk contiguity survive.
// Spreads half-wave fragment loads across all 64 LDS banks (4 hits/bank =
// the minimum for a 1KB ds_load_b128).
static __device__ __forceinline__ int swz64(int row, int e) {
  return ((((e >> 4) ^ row) & 3) << 4) | (e & 15);
}

#define TB_B 4
#define TB_S 1024
#define TB_D 512
#define TB_L 4
#define TB_H 8
#define TB_DK 64
#define TB_F 2048
#define TB_V 32000

// ---------------------------------------------------------------------------
// Embedding + sinusoidal positional encoding:  x = emb[tok]*sqrt(D) + PE
// ---------------------------------------------------------------------------
__global__ __launch_bounds__(256) void embed_kernel(
    const int* __restrict__ tokens, const float* __restrict__ emb,
    float* __restrict__ x, int S, int D) {
  int row = blockIdx.x;            // 0 .. B*S-1
  int s = row & (S - 1);
  int tok = tokens[row];
  const float sqrtD = 22.62741699796952f;  // sqrt(512)
  for (int d = threadIdx.x; d < D; d += blockDim.x) {
    int i = d >> 1;
    float freq = __expf((2.0f * (float)i) * (-logf(10000.0f) / (float)D));
    float ang = (float)s * freq;
    float pe = (d & 1) ? __cosf(ang) : __sinf(ang);
    x[(size_t)row * D + d] = emb[(size_t)tok * D + d] * sqrtD + pe;
  }
}

// ---------------------------------------------------------------------------
// LayerNorm (torch-style: eps added to std), bf16 output for matmul input
// ---------------------------------------------------------------------------
__global__ __launch_bounds__(256) void layernorm_kernel(
    const float* __restrict__ x, const float* __restrict__ gamma,
    const float* __restrict__ beta, bf16_t* __restrict__ out, int D) {
  int row = blockIdx.x;
  const float* xr = x + (size_t)row * D;
  __shared__ float red[8];
  int lane = threadIdx.x & 31, w = threadIdx.x >> 5;

  float s = 0.f;
  for (int i = threadIdx.x; i < D; i += 256) s += xr[i];
#pragma unroll
  for (int m = 16; m >= 1; m >>= 1) s += __shfl_xor(s, m, 32);
  if (lane == 0) red[w] = s;
  __syncthreads();
  float mean = 0.f;
#pragma unroll
  for (int i = 0; i < 8; ++i) mean += red[i];
  mean /= (float)D;
  __syncthreads();

  float vs = 0.f;
  for (int i = threadIdx.x; i < D; i += 256) {
    float d = xr[i] - mean;
    vs += d * d;
  }
#pragma unroll
  for (int m = 16; m >= 1; m >>= 1) vs += __shfl_xor(vs, m, 32);
  if (lane == 0) red[w] = vs;
  __syncthreads();
  float var = 0.f;
#pragma unroll
  for (int i = 0; i < 8; ++i) var += red[i];
  float inv = 1.0f / (sqrtf(var / (float)D) + 1e-6f);

  for (int i = threadIdx.x; i < D; i += 256)
    out[(size_t)row * D + i] = (bf16_t)(gamma[i] * (xr[i] - mean) * inv + beta[i]);
}

// ---------------------------------------------------------------------------
// Generic GEMM: C[M,N] = A_bf16[M,K] @ W_f32[K,N] + bias, with epilogues.
// BM=128, BN=128, BK=64; 256 threads = 8 waves (4x2), each wave -> 32x64.
// A tile staged via GLOBAL_LOAD_ASYNC_TO_LDS_B128 (ASYNCcnt) with per-lane
// permk+swizzle LDS destinations; W tile converted f32->bf16 in k-pairs.
// ---------------------------------------------------------------------------
#define EPI_F32_BIAS  0
#define EPI_F32_RES   1
#define EPI_BF16      2
#define EPI_BF16_GELU 3

template <int MODE>
__global__ __launch_bounds__(256) void gemm_bf16_kernel(
    const bf16_t* __restrict__ A, const float* __restrict__ W,
    const float* __restrict__ bias, const float* residual,
    float* outf, bf16_t* outb, int M, int N, int K) {
  __shared__ __align__(32) bf16_t As[128 * 64];  // [m][swz(permk(k))]
  __shared__ __align__(32) bf16_t Bs[128 * 64];  // N-major [n][swz(permk(k))]

  int tid = threadIdx.x;
  int lane = tid & 31, wave = tid >> 5;
  int hl = lane >> 4;                  // half-wave select
  int bm = blockIdx.x * 128;
  int bn = blockIdx.y * 128;
  int wm = (wave >> 1) * 32;           // 0,32,64,96
  int wn = (wave & 1) * 64;            // 0,64

  v8f acc[2][4];
#pragma unroll
  for (int i = 0; i < 2; ++i)
#pragma unroll
    for (int j = 0; j < 4; ++j)
#pragma unroll
      for (int v = 0; v < 8; ++v) acc[i][j][v] = 0.f;

  for (int k0 = 0; k0 < K; k0 += 64) {
    // --- prefetch next K-stage tiles into cache (global_prefetch_b8) ---
    if (k0 + 64 < K) {
      __builtin_prefetch(&A[(size_t)(bm + (tid >> 1)) * K + k0 + 64 + (tid & 1) * 32], 0, 1);
      __builtin_prefetch(&W[(size_t)(k0 + 64 + (tid >> 2)) * N + bn + (tid & 3) * 32], 0, 1);
    }
    // --- stage A tile (bf16, 128x64): async DMA global->LDS, permk+swizzle ---
#pragma unroll
    for (int i = 0; i < 4; ++i) {
      int c = i * 256 + tid;                 // 0..1023
      int r = c >> 3;                        // 0..127
      int col = (c & 7) * 8;                 // 0..56 (8-aligned)
      const bf16_t* gp = &A[(size_t)(bm + r) * K + k0 + col];
      bf16_t* lp = &As[r * 64 + swz64(r, permk(col))];
#if USE_ASYNC_LDS
      // LDS aperture maps addr[31:0] directly to the LDS byte offset.
      unsigned lds = (unsigned)(uintptr_t)lp;
      unsigned long long ga = (unsigned long long)(uintptr_t)gp;
      asm volatile("global_load_async_to_lds_b128 %0, %1, off"
                   :: "v"(lds), "v"(ga) : "memory");
#else
      *(v8bf*)lp = *(const v8bf*)gp;
#endif
    }
    // --- stage W tile (f32 -> bf16 k-pairs, N-major + swizzle) ---
#pragma unroll
    for (int i = 0; i < 16; ++i) {
      int idx = i * 256 + tid;               // 0..4095
      int kp = idx >> 7;                     // 0..31 (k pair)
      int n = idx & 127;                     // 0..127 (coalesced)
      float w0 = W[(size_t)(k0 + 2 * kp) * N + bn + n];
      float w1 = W[(size_t)(k0 + 2 * kp + 1) * N + bn + n];
      v2bf p;
      p[0] = (bf16_t)w0;
      p[1] = (bf16_t)w1;
      *(v2bf*)(&Bs[n * 64 + swz64(n, permk(2 * kp))]) = p;
    }
#if USE_ASYNC_LDS
    asm volatile("s_wait_asynccnt 0x0" ::: "memory");
#endif
    __syncthreads();

#pragma unroll
    for (int kc = 0; kc < 64; kc += 32) {
      v16bf afrag[2], bfrag[4];
#pragma unroll
      for (int t = 0; t < 2; ++t) {
        int r = wm + t * 16 + (lane & 15);
        afrag[t] = *(const v16bf*)(&As[r * 64 + swz64(r, kc + hl * 16)]);
      }
#pragma unroll
      for (int t = 0; t < 4; ++t) {
        int n = wn + t * 16 + (lane & 15);
        bfrag[t] = *(const v16bf*)(&Bs[n * 64 + swz64(n, kc + hl * 16)]);
      }
#pragma unroll
      for (int i = 0; i < 2; ++i)
#pragma unroll
        for (int j = 0; j < 4; ++j)
          acc[i][j] = wmma_bf16(afrag[i], bfrag[j], acc[i][j]);
    }
    __syncthreads();
  }

  // --- epilogue: straight-line, mode known at compile time ---
#pragma unroll
  for (int i = 0; i < 2; ++i) {
#pragma unroll
    for (int j = 0; j < 4; ++j) {
      int nn = bn + wn + j * 16 + (lane & 15);
      int m0 = bm + wm + i * 16 + hl * 8;
      float bv = bias[nn];
      size_t base = (size_t)m0 * N + nn;
#pragma unroll
      for (int v = 0; v < 8; ++v) {
        size_t off = base + (size_t)v * N;
        float val = acc[i][j][v] + bv;
        if constexpr (MODE == EPI_F32_BIAS) {
          outf[off] = val;
        } else if constexpr (MODE == EPI_F32_RES) {
          outf[off] = residual[off] + val;
        } else if constexpr (MODE == EPI_BF16) {
          outb[off] = (bf16_t)val;
        } else {  // exact GELU
          float g = 0.5f * val * (1.0f + erff(val * 0.7071067811865476f));
          outb[off] = (bf16_t)g;
        }
      }
    }
  }
}

// ---------------------------------------------------------------------------
// Flash attention (causal), bf16 Q/K/V, per (b, h, 64-query block).
// 128 threads = 4 waves; wave w owns query rows [qb+16w, qb+16w+16).
// Key blocks of 32; Q·K^T B-frags read K row-major from global directly.
// V staged transposed (swizzled LDS); P relayout via per-wave permk+swizzle.
// ---------------------------------------------------------------------------
__global__ __launch_bounds__(128) void attention_kernel(
    const bf16_t* __restrict__ Q, const bf16_t* __restrict__ Km,
    const bf16_t* __restrict__ V, bf16_t* __restrict__ O, int S) {
  const int D = TB_H * TB_DK;
  int qb = blockIdx.x * 64;
  int h = blockIdx.y;
  int b = blockIdx.z;
  int lane = threadIdx.x & 31, wave = threadIdx.x >> 5;
  int hl = lane >> 4;
  const size_t base = (size_t)b * S * D;
  const float scale = 0.125f;  // 1/sqrt(64)

  __shared__ __align__(32) bf16_t Vt[64 * 64];       // [dk][swz(permk(key))] (padded)
  __shared__ __align__(32) bf16_t Pb[4][16 * 64];    // per-wave P (padded)

  // Q fragments (16 rows x 64 dk = 2 A-frags), loaded once from global
  int qrow = qb + wave * 16 + (lane & 15);
  v16bf qfrag[2];
#pragma unroll
  for (int t = 0; t < 2; ++t) {
    const bf16_t* p = Q + base + (size_t)qrow * D + h * TB_DK + t * 32;
    v8bf lo = *(const v8bf*)(p + (hl ? 16 : 0));
    v8bf hi = *(const v8bf*)(p + 8 + (hl ? 16 : 0));
#pragma unroll
    for (int j = 0; j < 8; ++j) { qfrag[t][j] = lo[j]; qfrag[t][8 + j] = hi[j]; }
  }

  v8f oacc[4];
#pragma unroll
  for (int j = 0; j < 4; ++j)
#pragma unroll
    for (int v = 0; v < 8; ++v) oacc[j][v] = 0.f;
  float m_row[8], l_row[8];
#pragma unroll
  for (int v = 0; v < 8; ++v) { m_row[v] = -1e30f; l_row[v] = 0.f; }

  for (int kb = 0; kb < qb + 64; kb += 32) {
    // ---- scores: 16x32 tile = 2 n-tiles x 2 k-chunks of WMMA ----
    v8f sacc[2];
#pragma unroll
    for (int t = 0; t < 2; ++t)
#pragma unroll
      for (int v = 0; v < 8; ++v) sacc[t][v] = 0.f;
#pragma unroll
    for (int t = 0; t < 2; ++t) {
      int key = kb + t * 16 + (lane & 15);
#pragma unroll
      for (int kd = 0; kd < 2; ++kd) {
        // B-frag of K^T == contiguous row of K (bf16, 32B)
        v16bf kf = *(const v16bf*)(Km + base + (size_t)key * D + h * TB_DK +
                                   kd * 32 + (hl ? 16 : 0));
        sacc[t] = wmma_bf16(qfrag[kd], kf, sacc[t]);
      }
    }
    // ---- scale + causal mask + tile row max ----
    float tmax[8];
#pragma unroll
    for (int v = 0; v < 8; ++v) tmax[v] = -1e30f;
#pragma unroll
    for (int t = 0; t < 2; ++t) {
      int key = kb + t * 16 + (lane & 15);
#pragma unroll
      for (int v = 0; v < 8; ++v) {
        int qm = qb + wave * 16 + v + hl * 8;
        float sc = sacc[t][v] * scale;
        if (key > qm) sc = -1e9f;
        sacc[t][v] = sc;
        tmax[v] = fmaxf(tmax[v], sc);
      }
    }
#pragma unroll
    for (int v = 0; v < 8; ++v)
#pragma unroll
      for (int mm = 1; mm < 16; mm <<= 1)
        tmax[v] = fmaxf(tmax[v], __shfl_xor(tmax[v], mm, 32));
    // ---- online softmax update ----
    float alpha[8];
#pragma unroll
    for (int v = 0; v < 8; ++v) {
      float mn = fmaxf(m_row[v], tmax[v]);
      alpha[v] = __expf(m_row[v] - mn);
      m_row[v] = mn;
    }
    float tsum[8];
#pragma unroll
    for (int v = 0; v < 8; ++v) tsum[v] = 0.f;
#pragma unroll
    for (int t = 0; t < 2; ++t)
#pragma unroll
      for (int v = 0; v < 8; ++v) {
        float e = __expf(sacc[t][v] - m_row[v]);
        sacc[t][v] = e;
        tsum[v] += e;
      }
#pragma unroll
    for (int v = 0; v < 8; ++v)
#pragma unroll
      for (int mm = 1; mm < 16; mm <<= 1) tsum[v] += __shfl_xor(tsum[v], mm, 32);
#pragma unroll
    for (int v = 0; v < 8; ++v) l_row[v] = l_row[v] * alpha[v] + tsum[v];
#pragma unroll
    for (int j = 0; j < 4; ++j)
#pragma unroll
      for (int v = 0; v < 8; ++v) oacc[j][v] *= alpha[v];

    __syncthreads();  // previous iteration's Vt reads done
    // ---- stage V tile transposed: Vt[dk][key], 32 keys x 64 dk ----
#pragma unroll
    for (int i = 0; i < 4; ++i) {
      int c = i * 128 + threadIdx.x;  // 0..511 chunks of 4 bf16
      int key = c >> 4;
      int dk0 = (c & 15) * 4;
      const bf16_t* p = V + base + (size_t)(kb + key) * D + h * TB_DK + dk0;
#pragma unroll
      for (int j = 0; j < 4; ++j)
        Vt[(dk0 + j) * 64 + swz64(dk0 + j, permk(key))] = p[j];
    }
    // ---- store P (C-layout) to LDS permk+swizzled for A-layout reload ----
#pragma unroll
    for (int t = 0; t < 2; ++t)
#pragma unroll
      for (int v = 0; v < 8; ++v) {
        int m = v + hl * 8;
        int n = t * 16 + (lane & 15);
        Pb[wave][m * 64 + swz64(m, permk(n))] = (bf16_t)sacc[t][v];
      }
    __syncthreads();

    // ---- P @ V: A-frag from Pb (contiguous), 4 B-frags from Vt ----
    v16bf pf;
    {
      int r = lane & 15;
      pf = *(const v16bf*)(&Pb[wave][r * 64 + swz64(r, hl * 16)]);
    }
#pragma unroll
    for (int j = 0; j < 4; ++j) {
      int n = j * 16 + (lane & 15);
      v16bf vf = *(const v16bf*)(&Vt[n * 64 + swz64(n, hl * 16)]);
      oacc[j] = wmma_bf16(pf, vf, oacc[j]);
    }
  }

  // ---- normalize and write O as bf16 [B,S,D] ----
#pragma unroll
  for (int j = 0; j < 4; ++j) {
    int dk = j * 16 + (lane & 15);
    bf16_t* op = O + base + (size_t)(qb + wave * 16 + hl * 8) * D + h * TB_DK + dk;
#pragma unroll
    for (int v = 0; v < 8; ++v)
      op[(size_t)v * D] = (bf16_t)(oacc[j][v] / l_row[v]);
  }
}

// ---------------------------------------------------------------------------
// Host orchestration
// ---------------------------------------------------------------------------
static inline void launch_gemm(int mode, dim3 grid, hipStream_t stream,
                               const bf16_t* A, const float* W,
                               const float* bias, const float* res, float* of,
                               bf16_t* ob, int M, int N, int K) {
  switch (mode) {
    case EPI_F32_BIAS:
      gemm_bf16_kernel<EPI_F32_BIAS><<<grid, 256, 0, stream>>>(A, W, bias, res, of, ob, M, N, K);
      break;
    case EPI_F32_RES:
      gemm_bf16_kernel<EPI_F32_RES><<<grid, 256, 0, stream>>>(A, W, bias, res, of, ob, M, N, K);
      break;
    case EPI_BF16:
      gemm_bf16_kernel<EPI_BF16><<<grid, 256, 0, stream>>>(A, W, bias, res, of, ob, M, N, K);
      break;
    default:
      gemm_bf16_kernel<EPI_BF16_GELU><<<grid, 256, 0, stream>>>(A, W, bias, res, of, ob, M, N, K);
      break;
  }
}

extern "C" void kernel_launch(void* const* d_in, const int* in_sizes, int n_in,
                              void* d_out, int out_size, void* d_ws,
                              size_t ws_size, hipStream_t stream) {
  (void)in_sizes; (void)n_in; (void)out_size; (void)ws_size;
  const int* tokens   = (const int*)d_in[0];
  const float* tokemb = (const float*)d_in[1];
  const float* gamma1 = (const float*)d_in[2];
  const float* beta1  = (const float*)d_in[3];
  const float* wq = (const float*)d_in[4];  const float* bq = (const float*)d_in[5];
  const float* wk = (const float*)d_in[6];  const float* bk = (const float*)d_in[7];
  const float* wv = (const float*)d_in[8];  const float* bv = (const float*)d_in[9];
  const float* wo = (const float*)d_in[10]; const float* bo = (const float*)d_in[11];
  const float* gamma2 = (const float*)d_in[12];
  const float* beta2  = (const float*)d_in[13];
  const float* w1 = (const float*)d_in[14]; const float* b1 = (const float*)d_in[15];
  const float* w2 = (const float*)d_in[16]; const float* b2 = (const float*)d_in[17];
  const float* gammaf = (const float*)d_in[18];
  const float* betaf  = (const float*)d_in[19];
  const float* w_out = (const float*)d_in[20];
  const float* b_out = (const float*)d_in[21];

  const int B = TB_B, S = TB_S, D = TB_D, L = TB_L, F = TB_F, V = TB_V;
  const int M = B * S;  // 4096

  char* ws = (char*)d_ws;
  float*  x   = (float*)(ws);                       // 8 MB fp32 residual stream
  bf16_t* hbf = (bf16_t*)(ws + (8ull  << 20));      // 4 MB
  bf16_t* qbf = (bf16_t*)(ws + (12ull << 20));      // 4 MB
  bf16_t* kbf = (bf16_t*)(ws + (16ull << 20));      // 4 MB
  bf16_t* vbf = (bf16_t*)(ws + (20ull << 20));      // 4 MB
  bf16_t* obf = (bf16_t*)(ws + (24ull << 20));      // 4 MB
  bf16_t* gbf = (bf16_t*)(ws + (28ull << 20));      // 16 MB (gelu, M x F)

  embed_kernel<<<M, 256, 0, stream>>>(tokens, tokemb, x, S, D);

  dim3 gDD(M / 128, D / 128);   // 32 x 4
  dim3 gDF(M / 128, F / 128);   // 32 x 16
  dim3 gAttn(S / 64, TB_H, B);

  for (int l = 0; l < L; ++l) {
    size_t wDD = (size_t)l * D * D;
    size_t wDF = (size_t)l * D * F;
    layernorm_kernel<<<M, 256, 0, stream>>>(x, gamma1 + l * D, beta1 + l * D, hbf, D);
    launch_gemm(EPI_BF16, gDD, stream, hbf, wq + wDD, bq + l * D, nullptr,
                nullptr, qbf, M, D, D);
    launch_gemm(EPI_BF16, gDD, stream, hbf, wk + wDD, bk + l * D, nullptr,
                nullptr, kbf, M, D, D);
    launch_gemm(EPI_BF16, gDD, stream, hbf, wv + wDD, bv + l * D, nullptr,
                nullptr, vbf, M, D, D);
    attention_kernel<<<gAttn, 128, 0, stream>>>(qbf, kbf, vbf, obf, S);
    launch_gemm(EPI_F32_RES, gDD, stream, obf, wo + wDD, bo + l * D, x,
                x, nullptr, M, D, D);
    layernorm_kernel<<<M, 256, 0, stream>>>(x, gamma2 + l * D, beta2 + l * D, hbf, D);
    launch_gemm(EPI_BF16_GELU, gDF, stream, hbf, w1 + wDF, b1 + l * F, nullptr,
                nullptr, gbf, M, F, D);
    launch_gemm(EPI_F32_RES, gDD, stream, gbf, w2 + wDF, b2 + l * D, x,
                x, nullptr, M, D, F);
  }

  layernorm_kernel<<<M, 256, 0, stream>>>(x, gammaf, betaf, hbf, D);
  dim3 gOut(M / 128, V / 128);  // 32 x 250
  launch_gemm(EPI_F32_BIAS, gOut, stream, hbf, w_out, b_out, nullptr,
              (float*)d_out, nullptr, M, V, D);
}
